// SwitchingGRUDecoder_35244501631442
// MI455X (gfx1250) — matrix operand order repaired and checked by reading, not compiled, not run on hardware
//
#include <hip/hip_runtime.h>
#include <hip/hip_bf16.h>
#include <math.h>
#include <stdint.h>

typedef __attribute__((ext_vector_type(16))) _Float16     v16h;
typedef __attribute__((ext_vector_type(2)))  _Float16     h2;
typedef __attribute__((ext_vector_type(8)))  float        v8f;
typedef __attribute__((ext_vector_type(4)))  unsigned int u32x4;
typedef __attribute__((ext_vector_type(8)))  int          i32x8;
typedef __attribute__((ext_vector_type(4)))  int          i32x4;

#define S_DIM 50
#define B_DIM 64
#define T_DIM 32
#define E_DIM 512
#define H_DIM 1024
#define C_DIM 1024
#define V_DIM 32000

// ---------------------------------------------------------------------------
// Load 2 consecutive K-elements as packed f16 (from f32 with convert, or
// directly from f16).  Works for global and LDS (generic) pointers.
// ---------------------------------------------------------------------------
__device__ __forceinline__ h2 load2(const float* p)
{
    float2 v = *(const float2*)p;
    h2 r;
    r.x = (_Float16)v.x;
    r.y = (_Float16)v.y;
    return r;
}
__device__ __forceinline__ h2 load2(const _Float16* p)
{
    return *(const h2*)p;
}

// ---------------------------------------------------------------------------
// TDM: async-load one 2D weight panel (64 rows x 32 K-elements) into LDS.
// Descriptor per CDNA5 ISA 8.3/8.4: count=1, type=2 ("image"),
// data_size = sizeof(WT), tile_dim0=32, tile_dim1=64, row stride = ldw.
// Tile exactly matches tensor dims -> no OOB path.  Tracked by TENSORcnt.
// ---------------------------------------------------------------------------
template<typename WT>
__device__ __forceinline__ void tdm_load_panel(const WT* gsrc, unsigned lds_off,
                                               long ldw)
{
    const unsigned long long ga = (unsigned long long)(uintptr_t)gsrc;
    u32x4 g0;
    g0[0] = 1u;                                                 // count = 1
    g0[1] = lds_off;                                            // lds_addr
    g0[2] = (unsigned)ga;                                       // global_addr lo
    g0[3] = (unsigned)((ga >> 32) & 0x1FFFFFFu) | (2u << 30);   // addr hi | type=2

    const unsigned dsz = (sizeof(WT) == 2) ? 1u : 2u;           // 0=1B,1=2B,2=4B
    i32x8 g1;
    g1[0] = (int)(dsz << 16);            // wg_mask=0, data_size
    g1[1] = (int)(32u << 16);            // tensor_dim0[15:0] = 32
    g1[2] = (int)(64u << 16);            // tensor_dim0 hi=0, tensor_dim1[15:0]=64
    g1[3] = (int)(32u << 16);            // tensor_dim1 hi=0, tile_dim0=32
    g1[4] = 64;                          // tile_dim1=64, tile_dim2=0
    g1[5] = (int)(unsigned)ldw;          // tensor_dim0_stride (elements)
    g1[6] = 0;
    g1[7] = 0;

    i32x4 z4 = {0, 0, 0, 0};
#if defined(__clang_major__) && (__clang_major__ >= 23)
    i32x8 z8 = {0, 0, 0, 0, 0, 0, 0, 0};
    __builtin_amdgcn_tensor_load_to_lds(g0, g1, z4, z4, z8, 0);
#else
    __builtin_amdgcn_tensor_load_to_lds(g0, g1, z4, z4, 0);
#endif
}

// ---------------------------------------------------------------------------
// WMMA GEMM:  C[M,N] = act( A[M,K] @ W[N,K]^T + bias[N] )
// A row-major f32 (lda), W row-major (N,K) in WT (f32 or f16), K-contiguous.
// One wave per 32(M) x 64(N) register tile: 2 A-frags x 4 B-frags ->
// 8 v_wmma_f32_16x16x32_f16 per K-step (f32 accumulate).
// B panels double-buffered into private LDS via the Tensor Data Mover;
// A fragments loaded directly (prefetched).  Grid: (M/32, N/64), block = 32.
// Fragment layouts follow CDNA5 ISA 7.12.2 (wave32).  act: 0=none, 1=tanh.
// ---------------------------------------------------------------------------
template<typename WT>
__global__ __launch_bounds__(32) void wmma_gemm_nt(
    const float* __restrict__ A, int lda,
    const WT* __restrict__ W, int ldw,
    float* __restrict__ C, int ldc,
    int K,
    const float* __restrict__ bias, int act)
{
    __shared__ __align__(16) WT panel[2][64 * 32];

    const int tileM = blockIdx.x * 32;
    const int tileN = blockIdx.y * 64;
    const int lane  = threadIdx.x;
    const int half  = lane >> 4;      // 0: lanes 0-15, 1: lanes 16-31
    const int lrow  = lane & 15;

    const float* Ar[2];
    Ar[0] = A + (size_t)(tileM + lrow) * lda;
    Ar[1] = A + (size_t)(tileM + 16 + lrow) * lda;
    const WT* Wtile = W + (size_t)tileN * ldw;   // 64 weight rows, K-contiguous

    const unsigned lds0 = (unsigned)(uintptr_t)&panel[0][0];
    const unsigned lds1 = (unsigned)(uintptr_t)&panel[1][0];

    // prime the pipeline with the first B panel
    tdm_load_panel(Wtile, lds0, ldw);

    v8f acc[2][4] = {};
    int buf = 0;
    for (int k0 = 0; k0 < K; k0 += 32, buf ^= 1) {
        __builtin_prefetch(Ar[0] + k0 + 32, 0, 1);
        __builtin_prefetch(Ar[1] + k0 + 32, 0, 1);

        if (k0 + 32 < K) {
            tdm_load_panel(Wtile + (k0 + 32), buf ? lds0 : lds1, ldw);
            __builtin_amdgcn_s_wait_tensorcnt(1);   // current panel complete
        } else {
            __builtin_amdgcn_s_wait_tensorcnt(0);   // last panel complete
        }
        const WT* bp = &panel[buf][0];

        v16h a[2], b[4];
#pragma unroll
        for (int p = 0; p < 8; ++p) {
            // A-fragment 16x32: reg p holds K = (p&3)*2 + (p>>2)*16 + half*8 (+1)
            const int ka = k0 + ((p & 3) << 1) + ((p >> 2) << 4) + (half << 3);
            // B-fragment 32x16: reg p holds K = (p&3)*2 + (p>>2)*16 + half*8 (+1)
            const int kb = ((p & 3) << 1) + ((p >> 2) << 4) + (half << 3);
#pragma unroll
            for (int mi = 0; mi < 2; ++mi) {
                h2 t = load2(Ar[mi] + ka);
                a[mi][2 * p] = t.x; a[mi][2 * p + 1] = t.y;
            }
#pragma unroll
            for (int ni = 0; ni < 4; ++ni) {
                h2 t = load2(bp + (ni * 16 + lrow) * 32 + kb);
                b[ni][2 * p] = t.x; b[ni][2 * p + 1] = t.y;
            }
        }
#pragma unroll
        for (int mi = 0; mi < 2; ++mi)
#pragma unroll
            for (int ni = 0; ni < 4; ++ni)
                acc[mi][ni] = __builtin_amdgcn_wmma_f32_16x16x32_f16(
                    false, a[mi], false, b[ni], (short)0, acc[mi][ni], false, false);
    }

    // D layout: element (M = r + 8*half, N = lrow) lives in acc[r]
#pragma unroll
    for (int mi = 0; mi < 2; ++mi) {
#pragma unroll
        for (int ni = 0; ni < 4; ++ni) {
            const int col = tileN + ni * 16 + lrow;
            float bv = bias ? bias[col] : 0.0f;
#pragma unroll
            for (int r = 0; r < 8; ++r) {
                const int row = tileM + mi * 16 + r + (half << 3);
                float v = acc[mi][ni][r] + bv;
                if (act == 1) v = tanhf(v);
                C[(size_t)row * ldc + col] = v;
            }
        }
    }
}

// ---------------------------------------------------------------------------
__global__ void cvt_f32_to_f16(const float* __restrict__ src,
                               _Float16* __restrict__ dst, int n)
{
    int i = blockIdx.x * blockDim.x + threadIdx.x;
    if (i < n) dst[i] = (_Float16)src[i];
}

__global__ void embed_kernel(const int* __restrict__ y,
                             const float* __restrict__ emb,
                             float* __restrict__ y_emb, int rows)
{
    int i = blockIdx.x * blockDim.x + threadIdx.x;
    if (i >= rows * E_DIM) return;
    int r = i / E_DIM, e = i % E_DIM;
    int tok = y[r];
    y_emb[i] = (tok == 0) ? 0.0f : emb[(size_t)tok * E_DIM + e];
}

__global__ void zero_kernel(float* __restrict__ p, int n)
{
    int i = blockIdx.x * blockDim.x + threadIdx.x;
    if (i < n) p[i] = 0.0f;
}

// GRU pointwise: gate order r,z,n (torch GRUCell). gi,gh: (B,3H); h,hout: (B,H)
__global__ __launch_bounds__(256) void gru_pointwise(
    const float* __restrict__ gi, const float* __restrict__ gh,
    const float* __restrict__ h, float* __restrict__ hout)
{
    int idx = blockIdx.x * blockDim.x + threadIdx.x;
    int b = idx / H_DIM, j = idx % H_DIM;
    const float* gib = gi + (size_t)b * 3 * H_DIM;
    const float* ghb = gh + (size_t)b * 3 * H_DIM;
    float ir = gib[j],             hr = ghb[j];
    float iz = gib[H_DIM + j],     hz = ghb[H_DIM + j];
    float in = gib[2 * H_DIM + j], hn = ghb[2 * H_DIM + j];
    float r = 1.0f / (1.0f + __expf(-(ir + hr)));
    float z = 1.0f / (1.0f + __expf(-(iz + hz)));
    float n = tanhf(in + r * hn);
    hout[idx] = (1.0f - z) * n + z * h[idx];
}

// MLP attention, one block per batch element b.
__global__ __launch_bounds__(256) void attention_kernel(
    const float* __restrict__ ctx,       // (S,B,C)
    const float* __restrict__ ctx_proj,  // (S,B,C)
    const float* __restrict__ hidc,      // (B,C)
    const float* __restrict__ w_mlp,     // (C)
    float* __restrict__ zpre)            // (B,C)
{
    const int b = blockIdx.x, tid = threadIdx.x;
    __shared__ float red[256];
    __shared__ float sc[64];
    __shared__ float alpha[64];

    const float* hb = hidc + (size_t)b * C_DIM;
    for (int s = 0; s < S_DIM; ++s) {
        const float* cp = ctx_proj + ((size_t)s * B_DIM + b) * C_DIM;
        float p = 0.0f;
        for (int c = tid; c < C_DIM; c += 256)
            p += tanhf(cp[c] + hb[c]) * w_mlp[c];
        red[tid] = p;
        __syncthreads();
        for (int off = 128; off > 0; off >>= 1) {
            if (tid < off) red[tid] += red[tid + off];
            __syncthreads();
        }
        if (tid == 0) sc[s] = red[0];
        __syncthreads();
    }
    if (tid == 0) {
        float mx = sc[0];
        for (int s = 1; s < S_DIM; ++s) mx = fmaxf(mx, sc[s]);
        float sum = 0.0f;
        for (int s = 0; s < S_DIM; ++s) { alpha[s] = __expf(sc[s] - mx); sum += alpha[s]; }
        float inv = 1.0f / sum;
        for (int s = 0; s < S_DIM; ++s) alpha[s] *= inv;
    }
    __syncthreads();
    for (int c = tid; c < C_DIM; c += 256) {
        float acc = 0.0f;
        for (int s = 0; s < S_DIM; ++s)
            acc += alpha[s] * ctx[((size_t)s * B_DIM + b) * C_DIM + c];
        zpre[(size_t)b * C_DIM + c] = acc;
    }
}

// Single-pass online logsumexp + NLL (ignore_index = 0).  One block per b.
__global__ __launch_bounds__(256) void loss_kernel(
    const float* __restrict__ logits,   // (B, V)
    const int* __restrict__ y_next,     // (B,)
    float* __restrict__ losses)         // (B,)
{
    const int b = blockIdx.x, tid = threadIdx.x;
    const float* lb = logits + (size_t)b * V_DIM;
    __shared__ float rm[256];
    __shared__ float rs[256];

    float m = -3.4e38f, s = 0.0f;
    for (int v = tid; v < V_DIM; v += 256) {
        float x = lb[v];
        float nm = fmaxf(m, x);
        s = s * __expf(m - nm) + __expf(x - nm);
        m = nm;
    }
    rm[tid] = m; rs[tid] = s;
    __syncthreads();
    for (int off = 128; off > 0; off >>= 1) {
        if (tid < off) {
            float m1 = rm[tid], s1 = rs[tid];
            float m2 = rm[tid + off], s2 = rs[tid + off];
            float M = fmaxf(m1, m2);
            rs[tid] = s1 * __expf(m1 - M) + s2 * __expf(m2 - M);
            rm[tid] = M;
        }
        __syncthreads();
    }
    if (tid == 0) {
        int tgt = y_next[b];
        float lse = rm[0] + __logf(rs[0]);
        float nll = lse - lb[tgt];
        losses[b] = (tgt != 0) ? nll : 0.0f;
    }
}

__global__ __launch_bounds__(256) void sum_kernel(
    const float* __restrict__ losses, float* __restrict__ out, int n)
{
    __shared__ float red[256];
    float s = 0.0f;
    for (int i = threadIdx.x; i < n; i += 256) s += losses[i];
    red[threadIdx.x] = s; __syncthreads();
    for (int off = 128; off > 0; off >>= 1) {
        if (threadIdx.x < off) red[threadIdx.x] += red[threadIdx.x + off];
        __syncthreads();
    }
    if (threadIdx.x == 0) out[0] = red[0];
}

// ---------------------------------------------------------------------------
// Full decode pipeline, templated on weight storage type.
// ---------------------------------------------------------------------------
template<typename WT>
static void run_pipeline(
    const float* ctx, const int* y,
    const WT* W_ih0, const WT* W_hh0, const float* b_ih0, const float* b_hh0,
    const WT* W_ih1, const WT* W_hh1, const float* b_ih1, const float* b_hh1,
    const WT* W_c2c, const WT* W_h2c, const float* w_mlp, const WT* W_c2h,
    const WT* W_ho, const float* b_ho, const WT* W_op, const float* b_op,
    float* y_emb, float* gi0, float* ctx_proj, float* h, float* h1,
    float* gh0, float* hidc, float* zpre, float* zt, float* gi1, float* gh1,
    float* logit, float* logits_v, float* losses,
    float* d_out, hipStream_t stream)
{
    const int TT = T_DIM - 1;
    const int rows = TT * B_DIM;   // 1984

    wmma_gemm_nt<WT><<<dim3(rows / 32, 3 * H_DIM / 64), 32, 0, stream>>>(
        y_emb, E_DIM, W_ih0, E_DIM, gi0, 3 * H_DIM, E_DIM, b_ih0, 0);
    wmma_gemm_nt<WT><<<dim3(S_DIM * B_DIM / 32, C_DIM / 64), 32, 0, stream>>>(
        ctx, C_DIM, W_c2c, C_DIM, ctx_proj, C_DIM, C_DIM, nullptr, 0);
    zero_kernel<<<(B_DIM * H_DIM + 255) / 256, 256, 0, stream>>>(h, B_DIM * H_DIM);

    for (int t = 0; t < TT; ++t) {
        wmma_gemm_nt<WT><<<dim3(B_DIM / 32, 3 * H_DIM / 64), 32, 0, stream>>>(
            h, H_DIM, W_hh0, H_DIM, gh0, 3 * H_DIM, H_DIM, b_hh0, 0);
        gru_pointwise<<<B_DIM * H_DIM / 256, 256, 0, stream>>>(
            gi0 + (size_t)t * B_DIM * 3 * H_DIM, gh0, h, h1);
        wmma_gemm_nt<WT><<<dim3(B_DIM / 32, C_DIM / 64), 32, 0, stream>>>(
            h1, H_DIM, W_h2c, H_DIM, hidc, C_DIM, H_DIM, nullptr, 0);
        attention_kernel<<<B_DIM, 256, 0, stream>>>(ctx, ctx_proj, hidc, w_mlp, zpre);
        wmma_gemm_nt<WT><<<dim3(B_DIM / 32, H_DIM / 64), 32, 0, stream>>>(
            zpre, C_DIM, W_c2h, C_DIM, zt, H_DIM, C_DIM, nullptr, 0);
        wmma_gemm_nt<WT><<<dim3(B_DIM / 32, 3 * H_DIM / 64), 32, 0, stream>>>(
            zt, H_DIM, W_ih1, H_DIM, gi1, 3 * H_DIM, H_DIM, b_ih1, 0);
        wmma_gemm_nt<WT><<<dim3(B_DIM / 32, 3 * H_DIM / 64), 32, 0, stream>>>(
            h1, H_DIM, W_hh1, H_DIM, gh1, 3 * H_DIM, H_DIM, b_hh1, 0);
        gru_pointwise<<<B_DIM * H_DIM / 256, 256, 0, stream>>>(gi1, gh1, h1, h);
        wmma_gemm_nt<WT><<<dim3(B_DIM / 32, E_DIM / 64), 32, 0, stream>>>(
            h, H_DIM, W_ho, H_DIM, logit, E_DIM, H_DIM, b_ho, 1);
        wmma_gemm_nt<WT><<<dim3(B_DIM / 32, V_DIM / 64), 32, 0, stream>>>(
            logit, E_DIM, W_op, E_DIM, logits_v, V_DIM, E_DIM, b_op, 0);
        loss_kernel<<<B_DIM, 256, 0, stream>>>(logits_v, y + (size_t)(t + 1) * B_DIM,
                                               losses + (size_t)t * B_DIM);
    }
    sum_kernel<<<1, 256, 0, stream>>>(losses, d_out, TT * B_DIM);
}

// ---------------------------------------------------------------------------
extern "C" void kernel_launch(void* const* d_in, const int* in_sizes, int n_in,
                              void* d_out, int out_size, void* d_ws, size_t ws_size,
                              hipStream_t stream)
{
    const float* ctx   = (const float*)d_in[0];
    const int*   y     = (const int*)  d_in[1];
    const float* emb   = (const float*)d_in[2];
    const float* W_ih0 = (const float*)d_in[3];
    const float* W_hh0 = (const float*)d_in[4];
    const float* b_ih0 = (const float*)d_in[5];
    const float* b_hh0 = (const float*)d_in[6];
    const float* W_ih1 = (const float*)d_in[7];
    const float* W_hh1 = (const float*)d_in[8];
    const float* b_ih1 = (const float*)d_in[9];
    const float* b_hh1 = (const float*)d_in[10];
    const float* W_c2c = (const float*)d_in[11];
    const float* W_h2c = (const float*)d_in[12];
    const float* w_mlp = (const float*)d_in[13];
    const float* W_c2h = (const float*)d_in[14];
    const float* W_ho  = (const float*)d_in[15];
    const float* b_ho  = (const float*)d_in[16];
    const float* W_op  = (const float*)d_in[17];
    const float* b_op  = (const float*)d_in[18];
    (void)in_sizes; (void)n_in; (void)out_size;

    const int TT = T_DIM - 1;
    const int rows = TT * B_DIM;

    // ---- f32 workspace ----
    float* ws = (float*)d_ws;
    size_t off = 0;
    float* y_emb    = ws + off; off += (size_t)rows * E_DIM;
    float* gi0      = ws + off; off += (size_t)rows * 3 * H_DIM;
    float* ctx_proj = ws + off; off += (size_t)S_DIM * B_DIM * C_DIM;
    float* h        = ws + off; off += (size_t)B_DIM * H_DIM;
    float* h1       = ws + off; off += (size_t)B_DIM * H_DIM;
    float* gh0      = ws + off; off += (size_t)B_DIM * 3 * H_DIM;
    float* hidc     = ws + off; off += (size_t)B_DIM * C_DIM;
    float* zpre     = ws + off; off += (size_t)B_DIM * C_DIM;
    float* zt       = ws + off; off += (size_t)B_DIM * H_DIM;
    float* gi1      = ws + off; off += (size_t)B_DIM * 3 * H_DIM;
    float* gh1      = ws + off; off += (size_t)B_DIM * 3 * H_DIM;
    float* logit    = ws + off; off += (size_t)B_DIM * E_DIM;
    float* logits_v = ws + off; off += (size_t)B_DIM * V_DIM;
    float* losses   = ws + off; off += (size_t)TT * B_DIM;

    // ---- optional f16 weight cache (after f32 region) ----
    const size_t nW[9] = {
        (size_t)3 * H_DIM * E_DIM,  // W_ih0
        (size_t)3 * H_DIM * H_DIM,  // W_hh0
        (size_t)3 * H_DIM * H_DIM,  // W_ih1
        (size_t)3 * H_DIM * H_DIM,  // W_hh1
        (size_t)C_DIM * C_DIM,      // W_c2c
        (size_t)C_DIM * H_DIM,      // W_h2c
        (size_t)H_DIM * C_DIM,      // W_c2h
        (size_t)E_DIM * H_DIM,      // W_ho
        (size_t)V_DIM * E_DIM,      // W_op
    };
    size_t halfTotal = 0;
    for (int i = 0; i < 9; ++i) halfTotal += nW[i];
    const size_t bytesNeeded = off * sizeof(float) + halfTotal * sizeof(_Float16);

    embed_kernel<<<(rows * E_DIM + 255) / 256, 256, 0, stream>>>(y, emb, y_emb, rows);

    if (ws_size >= bytesNeeded) {
        _Float16* hbase = (_Float16*)(ws + off);
        const float* srcs[9] = {W_ih0, W_hh0, W_ih1, W_hh1, W_c2c, W_h2c, W_c2h, W_ho, W_op};
        _Float16* dsts[9];
        size_t hoff = 0;
        for (int i = 0; i < 9; ++i) {
            dsts[i] = hbase + hoff; hoff += nW[i];
            cvt_f32_to_f16<<<((int)nW[i] + 255) / 256, 256, 0, stream>>>(srcs[i], dsts[i], (int)nW[i]);
        }
        run_pipeline<_Float16>(ctx, y,
            dsts[0], dsts[1], b_ih0, b_hh0,
            dsts[2], dsts[3], b_ih1, b_hh1,
            dsts[4], dsts[5], w_mlp, dsts[6],
            dsts[7], b_ho, dsts[8], b_op,
            y_emb, gi0, ctx_proj, h, h1, gh0, hidc, zpre, zt, gi1, gh1,
            logit, logits_v, losses, (float*)d_out, stream);
    } else {
        run_pipeline<float>(ctx, y,
            W_ih0, W_hh0, b_ih0, b_hh0,
            W_ih1, W_hh1, b_ih1, b_hh1,
            W_c2c, W_h2c, w_mlp, W_c2h,
            W_ho, b_ho, W_op, b_op,
            y_emb, gi0, ctx_proj, h, h1, gh0, hidc, zpre, zt, gi1, gh1,
            logit, logits_v, losses, (float*)d_out, stream);
    }
}
